// CrossAttention_88270167867860
// MI455X (gfx1250) — matrix operand order repaired and checked by reading, not compiled, not run on hardware
//
#include <hip/hip_runtime.h>

// ---------------------------------------------------------------------------
// MI455X (gfx1250) dual-softmax bilinear attention, bf16 WMMA pipeline.
// B=4, N=3072, C=256, Cpad=288 (= 18 WMMA tiles), pos dims 6 -> C+6=262.
// Stats are stored pre-scaled by log2(e) so attn = exp2(2*L2E*a - Lr2 - Lc2)
// uses the native v_exp_f32 path; f32->bf16 uses hardware v_cvt_pk_bf16_f32.
// ---------------------------------------------------------------------------
#define NB 4
#define NN 3072
#define CC 256
#define CE 262
#define CP 288
#define HH 48
#define WW 64
#define L2E 1.4426950408889634f

typedef float          f4    __attribute__((ext_vector_type(4)));
typedef float          v8f   __attribute__((ext_vector_type(8)));
typedef unsigned int   u4    __attribute__((ext_vector_type(4)));
typedef unsigned int   u8v   __attribute__((ext_vector_type(8)));
typedef unsigned short v16us __attribute__((ext_vector_type(16)));
typedef __bf16         v16bf __attribute__((ext_vector_type(16)));

// Packed f32x2 -> bf16x2 via the hardware converter (1 VALU op per pair).
__device__ __forceinline__ unsigned pkbf(float lo, float hi) {
    unsigned r;
    asm("v_cvt_pk_bf16_f32 %0, %1, %2" : "=v"(r) : "v"(lo), "v"(hi));
    return r;
}
__device__ __forceinline__ unsigned short f2bf(float f) {
    return (unsigned short)(pkbf(f, 0.f) & 0xFFFFu);
}

__device__ __forceinline__ float fast_exp2(float x) {
#if __has_builtin(__builtin_amdgcn_exp2f)
    return __builtin_amdgcn_exp2f(x);
#else
    return exp2f(x);
#endif
}

__device__ __forceinline__ v8f wmma_bf16(v16bf a, v16bf b, v8f c) {
    // (neg_a, A, neg_b, B, c_mod, C, reuse_a, reuse_b)
    return __builtin_amdgcn_wmma_f32_16x16x32_bf16(false, a, false, b,
                                                   (short)0, c, false, false);
}

// Build v16bf A-fragment from two non-contiguous 8x bf16 (16B) chunks.
__device__ __forceinline__ v16bf comb(u4 lo, u4 hi) {
    u8v u;
    u[0] = lo[0]; u[1] = lo[1]; u[2] = lo[2]; u[3] = lo[3];
    u[4] = hi[0]; u[5] = hi[1]; u[6] = hi[2]; u[7] = hi[3];
    return __builtin_bit_cast(v16bf, u);
}

__device__ __forceinline__ v8f zero8() {
    v8f z = {0.f, 0.f, 0.f, 0.f, 0.f, 0.f, 0.f, 0.f};
    return z;
}

__device__ __forceinline__ float wave_rmax(float v) {
#pragma unroll
    for (int m = 16; m >= 1; m >>= 1) v = fmaxf(v, __shfl_xor(v, m, 32));
    return v;
}
__device__ __forceinline__ float wave_rsum(float v) {
#pragma unroll
    for (int m = 16; m >= 1; m >>= 1) v += __shfl_xor(v, m, 32);
    return v;
}

// exp2(fma(a, 2*log2e, base) - c) elementwise on float4 (packed-math friendly)
__device__ __forceinline__ f4 attn4(f4 a, f4 c, float base) {
    f4 k2 = {2.f * L2E, 2.f * L2E, 2.f * L2E, 2.f * L2E};
    f4 bb = {base, base, base, base};
    f4 s = bb - c;
    f4 t = __builtin_elementwise_fma(a, k2, s);
    f4 r;
#pragma unroll
    for (int q = 0; q < 4; ++q) r[q] = fast_exp2(t[q]);
    return r;
}

// ---------------------------------------------------------------------------
// 1) Row stats: Lr2[b][i] = (max_j a + log(sum_j exp(a-max))) * log2(e)
//    one wave32 per row, 8 waves per block.
// ---------------------------------------------------------------------------
__global__ void __launch_bounds__(256) stats_row_kernel(const float* __restrict__ a,
                                                        float* __restrict__ Lr) {
    int gw   = blockIdx.x * 8 + (threadIdx.x >> 5);
    int lane = threadIdx.x & 31;
    int b = gw / NN, i = gw - b * NN;
    const float* row = a + (size_t)b * NN * NN + (size_t)i * NN;
    float m = -3.402823466e38f;
    for (int j = lane; j < NN; j += 32) m = fmaxf(m, row[j]);
    m = wave_rmax(m);
    float s = 0.f;
    for (int j = lane; j < NN; j += 32) s += fast_exp2((row[j] - m) * L2E);
    s = wave_rsum(s);
    if (lane == 0) Lr[gw] = (m + __logf(s)) * L2E;
}

// ---------------------------------------------------------------------------
// 2) Column stats (scaled likewise): thread per column, coalesced row sweeps.
// ---------------------------------------------------------------------------
__global__ void __launch_bounds__(256) stats_col_kernel(const float* __restrict__ a,
                                                        float* __restrict__ Lc) {
    int b = blockIdx.x / (NN / 256);
    int j = (blockIdx.x % (NN / 256)) * 256 + threadIdx.x;
    const float* base = a + (size_t)b * NN * NN + j;
    float m = -3.402823466e38f;
    for (int i = 0; i < NN; ++i) m = fmaxf(m, base[(size_t)i * NN]);
    float s = 0.f;
    for (int i = 0; i < NN; ++i) s += fast_exp2((base[(size_t)i * NN] - m) * L2E);
    Lc[b * NN + j] = (m + __logf(s)) * L2E;
}

// ---------------------------------------------------------------------------
// 3) Build vT[b][c][n] (bf16, c padded to 288): LDS 32x32 tile transpose of x,
//    pos-encoding columns 256..261 computed analytically, 262..287 = 0.
// ---------------------------------------------------------------------------
__global__ void __launch_bounds__(256) prep_kernel(const float* __restrict__ x1,
                                                   const float* __restrict__ x2,
                                                   unsigned short* __restrict__ v1T,
                                                   unsigned short* __restrict__ v2T) {
    int which = blockIdx.z / NB;
    int b     = blockIdx.z % NB;
    const float* x = which ? x2 : x1;
    unsigned short* vT = which ? v2T : v1T;
    int n0 = blockIdx.x * 32, c0 = blockIdx.y * 32;
    int tx = threadIdx.x, ty = threadIdx.y;
    __shared__ float t[32][33];
    if (c0 < CC) {
#pragma unroll
        for (int r = 0; r < 4; ++r) {
            int n = n0 + ty + r * 8;
            t[ty + r * 8][tx] = x[(size_t)(b * NN + n) * CC + c0 + tx];
        }
    }
    __syncthreads();
#pragma unroll
    for (int r = 0; r < 4; ++r) {
        int c = c0 + ty + r * 8;
        int n = n0 + tx;
        float val;
        if (c < CC) {
            val = t[tx][ty + r * 8];
        } else {
            float p3 = -1.f + 2.f * (float)(n % HH) / (float)(HH - 1); // ys tiled
            float p4 = -1.f + 2.f * (float)(n / HH) / (float)(WW - 1); // xs repeated
            int k = c - CC;
            val = (k == 0) ? p3 * p3 : (k == 1) ? p4 * p4 : (k == 2) ? p3 * p4
                : (k == 3) ? p3 : (k == 4) ? p4 : (k == 5) ? 1.f : 0.f;
        }
        vT[((size_t)b * CP + c) * NN + n] = f2bf(val);
    }
}

// ---------------------------------------------------------------------------
// 4) T1T[d][i] = (attn @ v1)^T, attn built in-register from a + (Lr2,Lc2).
//    Wave: 16 output rows (i) x 9 d-tiles. A = attn (fp32 rows of a -> exp2 ->
//    packed bf16 cvt), B = v1T (contiguous bf16).
// ---------------------------------------------------------------------------
__global__ void __launch_bounds__(256) gemm_t1_kernel(const float* __restrict__ a,
                                                      const float* __restrict__ Lrg,
                                                      const float* __restrict__ Lcg,
                                                      const unsigned short* __restrict__ v1T,
                                                      unsigned short* __restrict__ T1T) {
    int b    = blockIdx.y;
    int w    = threadIdx.x >> 5, lane = threadIdx.x & 31;
    int m0   = blockIdx.x * 64 + (w >> 1) * 16;   // output row tile
    int dbase = (w & 1) * 144;                    // 9 of 18 d-tiles
    int l15 = lane & 15, h = lane >> 4;
    const float* lr = Lrg + b * NN;
    const float* lc = Lcg + b * NN;
    const unsigned short* V = v1T + (size_t)b * CP * NN;
    unsigned short* O = T1T + (size_t)b * CP * NN;

    int r = m0 + l15;
    float nLrr = -lr[r];                          // hoisted, pre-scaled
    const float* arow = a + (size_t)b * NN * NN + (size_t)r * NN;

    v8f acc[9];
#pragma unroll
    for (int t = 0; t < 9; ++t) acc[t] = zero8();

    for (int k0 = 0; k0 < NN; k0 += 32) {
        int j1 = k0 + h * 8;
        f4 a0 = *(const f4*)(arow + j1);
        f4 a1 = *(const f4*)(arow + j1 + 4);
        f4 a2 = *(const f4*)(arow + j1 + 16);
        f4 a3 = *(const f4*)(arow + j1 + 20);
        f4 c0 = *(const f4*)(lc + j1);
        f4 c1 = *(const f4*)(lc + j1 + 4);
        f4 c2 = *(const f4*)(lc + j1 + 16);
        f4 c3 = *(const f4*)(lc + j1 + 20);
        f4 e0 = attn4(a0, c0, nLrr);
        f4 e1 = attn4(a1, c1, nLrr);
        f4 e2 = attn4(a2, c2, nLrr);
        f4 e3 = attn4(a3, c3, nLrr);
        u8v uw;
        uw[0] = pkbf(e0[0], e0[1]);
        uw[1] = pkbf(e0[2], e0[3]);
        uw[2] = pkbf(e1[0], e1[1]);
        uw[3] = pkbf(e1[2], e1[3]);
        uw[4] = pkbf(e2[0], e2[1]);
        uw[5] = pkbf(e2[2], e2[3]);
        uw[6] = pkbf(e3[0], e3[1]);
        uw[7] = pkbf(e3[2], e3[3]);
        v16bf afrag = __builtin_bit_cast(v16bf, uw);
#pragma unroll
        for (int t = 0; t < 9; ++t) {
            int d = dbase + t * 16 + l15;
            const unsigned short* vp = V + (size_t)d * NN + k0 + h * 8;
            v16bf bfrag = comb(*(const u4*)vp, *(const u4*)(vp + 16));
            acc[t] = wmma_bf16(afrag, bfrag, acc[t]);
        }
    }
    // store transposed (T1T[d][i]) as bf16, 8 contiguous i per lane
#pragma unroll
    for (int t = 0; t < 9; ++t) {
        int d = dbase + t * 16 + l15;
        unsigned short* op = O + (size_t)d * NN + m0 + h * 8;
        u4 st;
#pragma unroll
        for (int q = 0; q < 4; ++q)
            st[q] = pkbf(acc[t][2 * q], acc[t][2 * q + 1]);
        *(u4*)op = st;
    }
}

// ---------------------------------------------------------------------------
// 5) T2T[d][j] = v2^T @ attn  (== (attn^T @ v2)^T). A = v2T (contiguous),
//    B = attn columns (strided fp32 reads of a, L2-resident).
//    Wave: one 16-col j-tile x 9 d-row tiles.
// ---------------------------------------------------------------------------
__global__ void __launch_bounds__(256) gemm_t2_kernel(const float* __restrict__ a,
                                                      const float* __restrict__ Lrg,
                                                      const float* __restrict__ Lcg,
                                                      const unsigned short* __restrict__ v2T,
                                                      unsigned short* __restrict__ T2T) {
    int b    = blockIdx.y;
    int w    = threadIdx.x >> 5, lane = threadIdx.x & 31;
    int j0   = blockIdx.x * 64 + (w >> 1) * 16;
    int cbase = (w & 1) * 144;                    // 9 of 18 d-row tiles
    int l15 = lane & 15, h = lane >> 4;
    const float* A = a + (size_t)b * NN * NN;
    const float* lr = Lrg + b * NN;
    const unsigned short* V = v2T + (size_t)b * CP * NN;
    unsigned short* O = T2T + (size_t)b * CP * NN;

    int j = j0 + l15;
    float nLcj = -Lcg[b * NN + j];                // hoisted, pre-scaled

    v8f acc[9];
#pragma unroll
    for (int t = 0; t < 9; ++t) acc[t] = zero8();

    for (int i0 = 0; i0 < NN; i0 += 32) {
        int ib = i0 + h * 16;
        f4 r0 = *(const f4*)(lr + ib);
        f4 r1 = *(const f4*)(lr + ib + 4);
        f4 r2 = *(const f4*)(lr + ib + 8);
        f4 r3 = *(const f4*)(lr + ib + 12);
        const float* ap = A + (size_t)ib * NN + j;
        f4 av0, av1, av2, av3;
#pragma unroll
        for (int q = 0; q < 4; ++q) {
            av0[q] = ap[(size_t)q * NN];
            av1[q] = ap[(size_t)(q + 4) * NN];
            av2[q] = ap[(size_t)(q + 8) * NN];
            av3[q] = ap[(size_t)(q + 12) * NN];
        }
        f4 e0 = attn4(av0, r0, nLcj);
        f4 e1 = attn4(av1, r1, nLcj);
        f4 e2 = attn4(av2, r2, nLcj);
        f4 e3 = attn4(av3, r3, nLcj);
        u8v uw;
        uw[0] = pkbf(e0[0], e0[1]);
        uw[1] = pkbf(e0[2], e0[3]);
        uw[2] = pkbf(e1[0], e1[1]);
        uw[3] = pkbf(e1[2], e1[3]);
        uw[4] = pkbf(e2[0], e2[1]);
        uw[5] = pkbf(e2[2], e2[3]);
        uw[6] = pkbf(e3[0], e3[1]);
        uw[7] = pkbf(e3[2], e3[3]);
        v16bf bfrag = __builtin_bit_cast(v16bf, uw);
#pragma unroll
        for (int t = 0; t < 9; ++t) {
            int c = cbase + t * 16 + l15;
            const unsigned short* vp = V + (size_t)c * NN + i0 + h * 16;
            u4 lo = *(const u4*)vp;
            u4 hi = *(const u4*)(vp + 8);
            acc[t] = wmma_bf16(comb(lo, hi), bfrag, acc[t]);
        }
    }
#pragma unroll
    for (int t = 0; t < 9; ++t) {
#pragma unroll
        for (int v = 0; v < 8; ++v) {
            int d = cbase + t * 16 + v + 8 * h;
            O[(size_t)d * NN + j] = f2bf(acc[t][v]);   // coalesced b16 stores
        }
    }
}

// ---------------------------------------------------------------------------
// 6) F[c][d] = sum_n v[n][c] * T[n][d]  (288x288, K=3072).
//    One wave per (c-tile, d-half); A = vT rows, B = TT rows, both contiguous.
// ---------------------------------------------------------------------------
__global__ void __launch_bounds__(32) gemm_f_kernel(const unsigned short* __restrict__ v1T,
                                                    const unsigned short* __restrict__ v2T,
                                                    const unsigned short* __restrict__ T1T,
                                                    const unsigned short* __restrict__ T2T,
                                                    float* __restrict__ F1,
                                                    float* __restrict__ F2) {
    int half  = blockIdx.x & 1;
    int ct    = blockIdx.x >> 1;     // 0..17
    int b     = blockIdx.y;
    int which = blockIdx.z;          // 0 -> F1, 1 -> F2
    const unsigned short* V = (which ? v2T : v1T) + (size_t)b * CP * NN;
    const unsigned short* T = (which ? T2T : T1T) + (size_t)b * CP * NN;
    float* F = (which ? F2 : F1) + (size_t)b * CP * CP;
    int lane = threadIdx.x & 31, l15 = lane & 15, h = lane >> 4;
    int c_row = ct * 16 + l15;
    int dbase = half * 144;

    v8f acc[9];
#pragma unroll
    for (int t = 0; t < 9; ++t) acc[t] = zero8();

    for (int n0 = 0; n0 < NN; n0 += 32) {
        const unsigned short* apf = V + (size_t)c_row * NN + n0 + h * 16;
        u4 lo = *(const u4*)apf;
        u4 hi = *(const u4*)(apf + 8);
        v16bf afrag = comb(lo, hi);
#pragma unroll
        for (int t = 0; t < 9; ++t) {
            int d = dbase + t * 16 + l15;
            const unsigned short* bp = T + (size_t)d * NN + n0 + h * 16;
            v16bf bfrag = __builtin_bit_cast(v16bf, *(const v16us*)bp);
            acc[t] = wmma_bf16(afrag, bfrag, acc[t]);
        }
    }
#pragma unroll
    for (int t = 0; t < 9; ++t) {
#pragma unroll
        for (int v = 0; v < 8; ++v) {
            int c = ct * 16 + v + 8 * h;
            int d = dbase + t * 16 + l15;
            F[(size_t)c * CP + d] = acc[t][v];
        }
    }
}

// ---------------------------------------------------------------------------
// 7) out[which][b][d][e] = b[e] + sum_c F[c][d] * W[c][e]; which 0 -> out2(F2).
// ---------------------------------------------------------------------------
__global__ void __launch_bounds__(256) proj_kernel(const float* __restrict__ F1,
                                                   const float* __restrict__ F2,
                                                   const float* __restrict__ Wp,
                                                   const float* __restrict__ bp,
                                                   float* __restrict__ out) {
    int d = blockIdx.x;        // 0..261
    int b = blockIdx.y;
    int which = blockIdx.z;    // 0 -> out2 from F2, 1 -> out1 from F1
    int e = threadIdx.x;
    const float* F = (which ? F1 : F2) + (size_t)b * CP * CP;
    float acc = bp[e];
    for (int c = 0; c < CE; ++c)
        acc = fmaf(F[(size_t)c * CP + d], Wp[c * CC + e], acc);
    out[(size_t)which * NB * CE * CC + ((size_t)b * CE + d) * CC + e] = acc;
}

// ---------------------------------------------------------------------------
extern "C" void kernel_launch(void* const* d_in, const int* in_sizes, int n_in,
                              void* d_out, int out_size, void* d_ws, size_t ws_size,
                              hipStream_t stream) {
    const float* x1   = (const float*)d_in[0];
    const float* x2   = (const float*)d_in[1];
    const float* corr = (const float*)d_in[2];
    const float* Wp   = (const float*)d_in[3];
    const float* bp   = (const float*)d_in[4];

    char* ws = (char*)d_ws;
    size_t off = 0;
    float* Lr = (float*)(ws + off); off += (size_t)NB * NN * 4;
    float* Lc = (float*)(ws + off); off += (size_t)NB * NN * 4;
    unsigned short* v1T = (unsigned short*)(ws + off); off += (size_t)NB * CP * NN * 2;
    unsigned short* v2T = (unsigned short*)(ws + off); off += (size_t)NB * CP * NN * 2;
    unsigned short* T1T = (unsigned short*)(ws + off); off += (size_t)NB * CP * NN * 2;
    unsigned short* T2T = (unsigned short*)(ws + off); off += (size_t)NB * CP * NN * 2;
    float* F1 = (float*)(ws + off); off += (size_t)NB * CP * CP * 4;
    float* F2 = (float*)(ws + off); off += (size_t)NB * CP * CP * 4;

    stats_row_kernel<<<NB * NN / 8, 256, 0, stream>>>(corr, Lr);
    stats_col_kernel<<<NB * (NN / 256), 256, 0, stream>>>(corr, Lc);
    prep_kernel<<<dim3(NN / 32, CP / 32, NB * 2), dim3(32, 8), 0, stream>>>(x1, x2, v1T, v2T);
    gemm_t1_kernel<<<dim3(NN / 64, NB), 256, 0, stream>>>(corr, Lr, Lc, v1T, T1T);
    gemm_t2_kernel<<<dim3(NN / 64, NB), 256, 0, stream>>>(corr, Lr, Lc, v2T, T2T);
    gemm_f_kernel<<<dim3(36, NB, 2), 32, 0, stream>>>(v1T, v2T, T1T, T2T, F1, F2);
    proj_kernel<<<dim3(CE, NB, 2), 256, 0, stream>>>(F1, F2, Wp, bp, (float*)d_out);
}
